// graphSAGE_71683004171208
// MI455X (gfx1250) — matrix operand order repaired and checked by reading, not compiled, not run on hardware
//
#include <hip/hip_runtime.h>
#include <hip/hip_bf16.h>
#include <math.h>

// GraphSAGE 2-layer forward for MI455X (gfx1250, wave32, WMMA).
//
//   hid = relu( segsum((x@Wl1)[src]) / deg + b1 + x@Wr1 )   (transform-then-scatter)
//   out = log_softmax( segsum((hid@Wl2)[src]) / deg + b2 + hid@Wr2 )

#define NN   100000
#define EE   1600000
#define INC  128
#define HIDC 64
#define OUTC 40

typedef __attribute__((ext_vector_type(16))) __bf16 v16bf;
typedef __attribute__((ext_vector_type(8)))  float  v8f;

// ---------------------------------------------------------------------------
// Dual GEMM: Y = X @ Wl, Z = X @ Wr   (X: nrows x K f32, W: K x C f32, row-major)
// One 16-row tile per block; wave w computes col tile [16w, 16w+16).
// bf16 inputs, f32 accumulation via v_wmma_f32_16x16x32_bf16.
// No predicated loads: rows are exact multiples of 16 (clamped branch-free),
// out-of-range columns load a clamped (finite, in-bounds) W value and their
// results are simply never stored.
// ---------------------------------------------------------------------------
template <int K, int C>
__global__ void sage_dual_gemm(const float* __restrict__ X,
                               const float* __restrict__ Wl,
                               const float* __restrict__ Wr,
                               float* __restrict__ Y,
                               float* __restrict__ Z,
                               int nrows) {
  const int lane    = threadIdx.x & 31;
  const int wave    = threadIdx.x >> 5;
  const int half    = (lane >> 4) & 1;   // 0: lanes 0-15, 1: lanes 16-31
  const int l15     = lane & 15;
  const int rowBase = blockIdx.x * 16;
  const int colBase = wave * 16;

  int arow = rowBase + l15;              // A row for this lane
  arow = (arow < nrows) ? arow : (nrows - 1);   // branch-free clamp
  const int col  = colBase + l15;        // B/D column for this lane
  const int bcol = (col < C) ? col : (C - 1);   // clamped load column

  const float* __restrict__ arowp = X + (size_t)arow * K;

  v8f accL = {};
  v8f accR = {};

#pragma unroll
  for (int kt = 0; kt < K; kt += 32) {
    // ---- A tile 16x32 (ISA 7.12.2 16-bit A layout):
    //   lanes 0-15 : a[0..7] = K kt+0..7,  a[8..15] = K kt+16..23
    //   lanes 16-31: a[0..7] = K kt+8..15, a[8..15] = K kt+24..31
    // Two contiguous 8-float runs per lane -> four global_load_b128.
    const int kb = kt + 8 * half;
    float4 x0 = *(const float4*)(arowp + kb);
    float4 x1 = *(const float4*)(arowp + kb + 4);
    float4 x2 = *(const float4*)(arowp + kb + 16);
    float4 x3 = *(const float4*)(arowp + kb + 20);

    // ---- B tiles 32x16: lane = column; lanes 0-15 hold K kt+0..7,16..23,
    // lanes 16-31 hold K kt+8..15,24..31. Stage into f32 arrays so the compiler
    // issues all loads before one wait, then pack-convert.
    float blf[16], brf[16];
#pragma unroll
    for (int i = 0; i < 16; ++i) {
      int k = kt + (i < 8 ? i : i + 8) + (half ? 8 : 0);
      blf[i] = Wl[(size_t)k * C + bcol];
      brf[i] = Wr[(size_t)k * C + bcol];
    }

    v16bf a, bl, br;
    a[0] = (__bf16)x0.x; a[1] = (__bf16)x0.y; a[2] = (__bf16)x0.z; a[3] = (__bf16)x0.w;
    a[4] = (__bf16)x1.x; a[5] = (__bf16)x1.y; a[6] = (__bf16)x1.z; a[7] = (__bf16)x1.w;
    a[8]  = (__bf16)x2.x; a[9]  = (__bf16)x2.y; a[10] = (__bf16)x2.z; a[11] = (__bf16)x2.w;
    a[12] = (__bf16)x3.x; a[13] = (__bf16)x3.y; a[14] = (__bf16)x3.z; a[15] = (__bf16)x3.w;
#pragma unroll
    for (int i = 0; i < 16; ++i) {
      bl[i] = (__bf16)blf[i];
      br[i] = (__bf16)brf[i];
    }

    accL = __builtin_amdgcn_wmma_f32_16x16x32_bf16(false, a, false, bl,
                                                   (short)0, accL, false, false);
    accR = __builtin_amdgcn_wmma_f32_16x16x32_bf16(false, a, false, br,
                                                   (short)0, accR, false, false);
  }

  // D layout: col = colBase + (lane&15); row = rowBase + v + 8*half
  if (col < C) {
#pragma unroll
    for (int v = 0; v < 8; ++v) {
      int r = rowBase + v + (half ? 8 : 0);
      if (r < nrows) {
        Y[(size_t)r * C + col] = accL[v];
        Z[(size_t)r * C + col] = accR[v];
      }
    }
  }
}

// ---------------------------------------------------------------------------
// Degree: deg[dst] += 1 per edge
// ---------------------------------------------------------------------------
__global__ void degree_kernel(const int* __restrict__ ei, float* __restrict__ deg) {
  int e = blockIdx.x * blockDim.x + threadIdx.x;
  if (e < EE) atomicAdd(&deg[ei[EE + e]], 1.0f);
}

// ---------------------------------------------------------------------------
// Scatter-add, width 64: agg[dst] += y[src]. 64 lanes per edge -> coalesced.
// ---------------------------------------------------------------------------
__global__ void scatter64(const int* __restrict__ ei, const float* __restrict__ y,
                          float* __restrict__ agg) {
  int e = blockIdx.x * 4 + (threadIdx.x >> 6);
  int f = threadIdx.x & 63;
  if (e < EE) {
    int s = ei[e];
    int d = ei[EE + e];
    atomicAdd(&agg[(size_t)d * 64 + f], y[(size_t)s * 64 + f]);
  }
}

// ---------------------------------------------------------------------------
// Scatter-add, width 40 (flattened E*40 index space)
// ---------------------------------------------------------------------------
__global__ void scatter40(const int* __restrict__ ei, const float* __restrict__ y,
                          float* __restrict__ agg) {
  long long idx = (long long)blockIdx.x * blockDim.x + threadIdx.x;
  if (idx < (long long)EE * OUTC) {
    int e = (int)(idx / OUTC);
    int f = (int)(idx % OUTC);
    int s = ei[e];
    int d = ei[EE + e];
    atomicAdd(&agg[(size_t)d * OUTC + f], y[(size_t)s * OUTC + f]);
  }
}

// ---------------------------------------------------------------------------
// hid = relu(agg/deg + bias + z), width 64, float4-vectorized
// ---------------------------------------------------------------------------
__global__ void combine_relu64(const float4* __restrict__ agg, const float4* __restrict__ z,
                               const float* __restrict__ bias, const float* __restrict__ deg,
                               float4* __restrict__ out) {
  int idx = blockIdx.x * blockDim.x + threadIdx.x;   // over N*16 float4s
  if (idx < NN * 16) {
    int i  = idx >> 4;        // node
    int f4 = (idx & 15) << 2; // first feature of this float4
    float d = fmaxf(deg[i], 1.0f);
    float rd = __frcp_rn(d);
    float4 a = agg[idx];
    float4 w = z[idx];
    float4 r;
    r.x = fmaxf(a.x * rd + bias[f4 + 0] + w.x, 0.0f);
    r.y = fmaxf(a.y * rd + bias[f4 + 1] + w.y, 0.0f);
    r.z = fmaxf(a.z * rd + bias[f4 + 2] + w.z, 0.0f);
    r.w = fmaxf(a.w * rd + bias[f4 + 3] + w.w, 0.0f);
    out[idx] = r;
  }
}

// ---------------------------------------------------------------------------
// out = log_softmax(agg/deg + bias + z), width 40. One wave32 per row:
// lane covers f = lane and (lane<8) f = lane+32; shuffle reductions.
// ---------------------------------------------------------------------------
__global__ void combine_logsoftmax40(const float* __restrict__ agg, const float* __restrict__ z,
                                     const float* __restrict__ bias, const float* __restrict__ deg,
                                     float* __restrict__ out) {
  int lane = threadIdx.x & 31;
  int row  = blockIdx.x * (blockDim.x >> 5) + (threadIdx.x >> 5);
  if (row >= NN) return;

  float d = fmaxf(deg[row], 1.0f);
  size_t base = (size_t)row * OUTC;

  float v0 = agg[base + lane] / d + bias[lane] + z[base + lane];
  float v1 = -INFINITY;
  if (lane < 8) {
    int f = lane + 32;
    v1 = agg[base + f] / d + bias[f] + z[base + f];
  }

  float m = fmaxf(v0, v1);
#pragma unroll
  for (int off = 16; off > 0; off >>= 1) m = fmaxf(m, __shfl_xor(m, off, 32));

  float s = __expf(v0 - m) + ((lane < 8) ? __expf(v1 - m) : 0.0f);
#pragma unroll
  for (int off = 16; off > 0; off >>= 1) s += __shfl_xor(s, off, 32);

  float lse = m + __logf(s);
  out[base + lane] = v0 - lse;
  if (lane < 8) out[base + lane + 32] = v1 - lse;
}

// ---------------------------------------------------------------------------
extern "C" void kernel_launch(void* const* d_in, const int* in_sizes, int n_in,
                              void* d_out, int out_size, void* d_ws, size_t ws_size,
                              hipStream_t stream) {
  (void)in_sizes; (void)n_in; (void)out_size; (void)ws_size;

  const float* x   = (const float*)d_in[0];
  const int*   ei  = (const int*)d_in[1];      // [2, E]: row 0 = src, row 1 = dst
  const float* Wl1 = (const float*)d_in[2];
  const float* b1  = (const float*)d_in[3];
  const float* Wr1 = (const float*)d_in[4];
  const float* Wl2 = (const float*)d_in[5];
  const float* b2  = (const float*)d_in[6];
  const float* Wr2 = (const float*)d_in[7];
  float*       out = (float*)d_out;

  // Workspace carve-up (all f32): 4 buffers of N*64 + deg  (~103 MB)
  float* ws  = (float*)d_ws;
  size_t n64 = (size_t)NN * 64;
  float* y   = ws;            // X@Wl (layer 1: N x 64; layer 2 reuse: N x 40)
  float* z   = ws + n64;      // X@Wr
  float* agg = ws + 2 * n64;  // scatter accumulator
  float* hid = ws + 3 * n64;  // layer-1 output
  float* deg = ws + 4 * n64;  // N degrees

  const int rowTiles = (NN + 15) / 16;   // 6250 (exact)

  // -------- degree (shared by both layers) --------
  hipMemsetAsync(deg, 0, (size_t)NN * sizeof(float), stream);
  degree_kernel<<<(EE + 255) / 256, 256, 0, stream>>>(ei, deg);

  // -------- layer 1 --------
  hipMemsetAsync(agg, 0, n64 * sizeof(float), stream);
  sage_dual_gemm<INC, HIDC><<<rowTiles, 4 * 32, 0, stream>>>(x, Wl1, Wr1, y, z, NN);
  scatter64<<<(EE + 3) / 4, 256, 0, stream>>>(ei, y, agg);
  combine_relu64<<<(NN * 16 + 255) / 256, 256, 0, stream>>>(
      (const float4*)agg, (const float4*)z, b1, deg, (float4*)hid);

  // -------- layer 2 (reuse y/z/agg at width 40) --------
  hipMemsetAsync(agg, 0, (size_t)NN * OUTC * sizeof(float), stream);
  sage_dual_gemm<HIDC, OUTC><<<rowTiles, 3 * 32, 0, stream>>>(hid, Wl2, Wr2, y, z, NN);
  {
    long long total = (long long)EE * OUTC;
    int blocks = (int)((total + 255) / 256);
    scatter40<<<blocks, 256, 0, stream>>>(ei, y, agg);
  }
  combine_logsoftmax40<<<(NN + 7) / 8, 256, 0, stream>>>(agg, z, b2, deg, out);
}
